// KGNN_8684423873312
// MI455X (gfx1250) — matrix operand order repaired and checked by reading, not compiled, not run on hardware
//
#include <hip/hip_runtime.h>
#include <hip/hip_bf16.h>

typedef __bf16 bf16;
typedef __attribute__((ext_vector_type(16))) __bf16 v16bf;
typedef __attribute__((ext_vector_type(8)))  __bf16 v8bf;
typedef __attribute__((ext_vector_type(8)))  float  v8f;

#define HID 128
#define LDA 136   // LDS row stride (bf16 elems): multiple of 8, offsets banks by 4/row
#define GEMM_MAX_BLOCKS 2048

// ---------------- small utility kernels ----------------

__global__ void zero_f32_kernel(float* __restrict__ p, long n) {
    long i = (long)blockIdx.x * blockDim.x + threadIdx.x;
    long stride = (long)gridDim.x * blockDim.x;
    for (; i < n; i += stride) p[i] = 0.0f;
}

__global__ void cvt_bf16_kernel(bf16* __restrict__ dst, const float* __restrict__ src, int n) {
    int i = blockIdx.x * blockDim.x + threadIdx.x;
    if (i < n) dst[i] = (bf16)src[i];
}

// Build combined edge-head weight: rows 0..63 = edge_w[:, :128], rows 64..127 = edge_w[:, 128:]
__global__ void build_wcomb_kernel(bf16* __restrict__ dst, const float* __restrict__ edge_w) {
    int i = blockIdx.x * blockDim.x + threadIdx.x;   // 0..16383
    if (i >= 128 * HID) return;
    int n = i >> 7, k = i & 127;
    dst[i] = (bf16)edge_w[(n & 63) * 256 + (n >> 6) * 128 + k];
}

// rel_proj[r][c] = dot(rel_emb[r], lin_w[c]) + lin_b[c]   (64 x 128, tiny)
__global__ void rel_proj_kernel(float* __restrict__ out, const float* __restrict__ rel_emb,
                                const float* __restrict__ lin_w, const float* __restrict__ lin_b) {
    int r = blockIdx.x;
    int c = threadIdx.x;
    const float* e = rel_emb + r * HID;
    const float* w = lin_w + c * HID;
    float acc = lin_b[c];
    #pragma unroll 8
    for (int k = 0; k < HID; ++k) acc = fmaf(e[k], w[k], acc);
    out[r * HID + c] = acc;
}

// ---------------- WMMA GEMM: out[M,Nout] = (a0[rows] (+a1[rows])) @ Wb^T (+bias) ----------------
// Wb is bf16 [Nout, 128] row-major (so Wb row n = column n of the B matrix).
// Persistent blocks: 256 threads = 8 waves; each wave owns one 16-wide column tile and keeps its
// B fragments in registers; the block grid-strides over 16-row tiles of A.
// REQUIREMENT: M % 16 == 0 (true for all launches here: 200000, 256).
template <bool HAS_A1, bool HAS_IDX, bool HAS_BIAS, bool RELU>
__launch_bounds__(256)
__global__ void wmma_gemm_kernel(const float* __restrict__ a0,
                                 const float* __restrict__ a1,       // addend (same rows), if HAS_A1
                                 const int*   __restrict__ row_idx,  // row gather, if HAS_IDX
                                 const bf16*  __restrict__ wb,       // [Nout, 128] bf16
                                 const float* __restrict__ bias,     // [Nout], if HAS_BIAS
                                 float* __restrict__ out,            // [M, Nout]
                                 int M, int Nout) {
    __shared__ __align__(16) bf16 lds_a[16 * LDA];

    const int nTiles   = M >> 4;
    const int colBlock = blockIdx.y * 128;

    const int wave = threadIdx.x >> 5;
    const int lane = threadIdx.x & 31;
    const int colTile = colBlock + wave * 16;
    const bool active = colTile < Nout;

    const int halfsel = lane >> 4;   // 0: K{0-15}; 1: K{16-31}
    const int lrow    = lane & 15;   // A row within tile / output column within tile
    const int col     = colTile + lrow;

    // Preload this wave's B fragments once (reused for every row tile).
    v16bf bfrag[4];
    if (active) {
        const bf16* wbp = wb + (long)col * HID + halfsel * 16;
        #pragma unroll
        for (int kc = 0; kc < 4; ++kc)
            bfrag[kc] = *(const v16bf*)(wbp + kc * 32);
    }

    // Staging coordinates for this thread.
    const int sr  = threadIdx.x >> 4;        // 0..15
    const int sc0 = (threadIdx.x & 15) * 8;  // 0,8,...,120
    const float bv = (HAS_BIAS && active) ? bias[col] : 0.0f;

    for (int tile = blockIdx.x; tile < nTiles; tile += gridDim.x) {
        const int rowBase = tile * 16;

        // ---- stage A tile (16 x 128 f32 -> bf16 in LDS) ----
        {
            const int grow = rowBase + sr;
            const int arow = HAS_IDX ? row_idx[grow] : grow;
            const float* p0 = a0 + (long)arow * HID + sc0;
            float vals[8];
            #pragma unroll
            for (int j = 0; j < 8; ++j) vals[j] = p0[j];
            if (HAS_A1) {
                const float* p1 = a1 + (long)arow * HID + sc0;
                #pragma unroll
                for (int j = 0; j < 8; ++j) vals[j] += p1[j];
            }
            #pragma unroll
            for (int j = 0; j < 8; ++j) lds_a[sr * LDA + sc0 + j] = (bf16)vals[j];
        }
        __syncthreads();

        if (active) {
            v8f acc = {};
            #pragma unroll
            for (int kc = 0; kc < 4; ++kc) {
                // A fragment: 16 contiguous K per lane (k = 16*halfsel + s), matching B
                const bf16* abase = &lds_a[lrow * LDA + kc * 32 + halfsel * 16];
                v8bf alo = *(const v8bf*)abase;
                v8bf ahi = *(const v8bf*)(abase + 8);
                v16bf afrag;
                #pragma unroll
                for (int j = 0; j < 8; ++j) { afrag[j] = alo[j]; afrag[8 + j] = ahi[j]; }
                acc = __builtin_amdgcn_wmma_f32_16x16x32_bf16(false, afrag, false, bfrag[kc],
                                                              (short)0, acc, false, false);
            }
            // D layout: vgpr j -> row = rowBase + j + 8*halfsel, column = col (coalesced per row)
            float* po = out + (long)(rowBase + halfsel * 8) * Nout + col;
            #pragma unroll
            for (int j = 0; j < 8; ++j) {
                float v = acc[j];
                if (HAS_BIAS) v += bv;
                if (RELU)     v = fmaxf(v, 0.0f);
                po[(long)j * Nout] = v;
            }
        }
        __syncthreads();   // LDS reads done before next tile's staging overwrites it
    }
}

// ---------------- edge message scatter: agg[dst] += relu(x[src] + rel_proj[rel]) ----------------
// one wave per edge; lane handles 4 contiguous channels (float4)
__launch_bounds__(256)
__global__ void edge_scatter_kernel(const float* __restrict__ x,
                                    const float* __restrict__ rel_proj,
                                    const int* __restrict__ edge_index,  // [2,E] flat
                                    const int* __restrict__ rel_ids,
                                    float* __restrict__ agg, int E) {
    int wave = threadIdx.x >> 5;
    int lane = threadIdx.x & 31;
    int e = blockIdx.x * 8 + wave;
    if (e >= E) return;
    int src = edge_index[e];
    int dst = edge_index[E + e];
    int rel = rel_ids[e];
    int c = lane * 4;
    float4 xv = *(const float4*)(x + (long)src * HID + c);
    float4 rv = *(const float4*)(rel_proj + (long)rel * HID + c);
    float4 m;
    m.x = fmaxf(xv.x + rv.x, 0.0f);
    m.y = fmaxf(xv.y + rv.y, 0.0f);
    m.z = fmaxf(xv.z + rv.z, 0.0f);
    m.w = fmaxf(xv.w + rv.w, 0.0f);
    float* dp = agg + (long)dst * HID + c;
    atomicAdd(dp + 0, m.x);
    atomicAdd(dp + 1, m.y);
    atomicAdd(dp + 2, m.z);
    atomicAdd(dp + 3, m.w);
}

// ---------------- edge head: out[e][r] = P[src][r] + P[dst][64+r] + edge_b[r] ----------------
__launch_bounds__(256)
__global__ void edge_head_kernel(const float* __restrict__ P,
                                 const int* __restrict__ edge_index,
                                 const float* __restrict__ edge_b,
                                 float* __restrict__ out, int E) {
    int wave = threadIdx.x >> 5;
    int lane = threadIdx.x & 31;
    int e = blockIdx.x * 8 + wave;
    if (e >= E) return;
    int src = edge_index[e];
    int dst = edge_index[E + e];
    int r = lane * 2;
    float2 a  = *(const float2*)(P + (long)src * HID + r);
    float2 b  = *(const float2*)(P + (long)dst * HID + 64 + r);
    float2 eb = *(const float2*)(edge_b + r);
    float2 o;
    o.x = a.x + b.x + eb.x;
    o.y = a.y + b.y + eb.y;
    *(float2*)(out + (long)e * 64 + r) = o;
}

// ---------------- launch ----------------

extern "C" void kernel_launch(void* const* d_in, const int* in_sizes, int n_in,
                              void* d_out, int out_size, void* d_ws, size_t ws_size,
                              hipStream_t stream) {
    (void)n_in; (void)out_size; (void)ws_size;
    const float* node_emb = (const float*)d_in[0];
    const float* rel_emb  = (const float*)d_in[1];
    const float* lin_w    = (const float*)d_in[2];
    const float* lin_b    = (const float*)d_in[3];
    const float* conv1_w  = (const float*)d_in[4];
    const float* conv1_b  = (const float*)d_in[5];
    const float* conv2_w  = (const float*)d_in[6];
    const float* conv2_b  = (const float*)d_in[7];
    const float* edge_w   = (const float*)d_in[8];
    const float* edge_b   = (const float*)d_in[9];
    const float* motif_w  = (const float*)d_in[10];
    const float* motif_b  = (const float*)d_in[11];
    const float* node_w   = (const float*)d_in[12];
    const float* node_b   = (const float*)d_in[13];
    const int* node_ids   = (const int*)d_in[14];
    const int* rel_ids    = (const int*)d_in[15];
    const int* center_idx = (const int*)d_in[16];
    const int* edge_index = (const int*)d_in[17];

    const int N       = in_sizes[14];          // 200000
    const int E       = in_sizes[15];          // 600000
    const int C       = in_sizes[16];          // 256
    const int NRELS   = in_sizes[9];           // 64
    const int NMOTIF  = in_sizes[11];          // 512
    const int NCLS    = in_sizes[13];          // 16

    // workspace carve-up (256B aligned slices)
    char*  ws  = (char*)d_ws;
    size_t off = 0;
    auto carve = [&](size_t bytes) -> void* {
        void* p = ws + off;
        off = (off + bytes + 255) & ~(size_t)255;
        return p;
    };
    float* xbuf    = (float*)carve((size_t)N * HID * sizeof(float));   // node features
    float* agg     = (float*)carve((size_t)N * HID * sizeof(float));   // agg, later reused as P
    float* relp    = (float*)carve((size_t)NRELS * HID * sizeof(float));
    bf16*  wb_lin  = (bf16*)carve((size_t)HID * HID * sizeof(bf16));
    bf16*  wb_c1   = (bf16*)carve((size_t)HID * HID * sizeof(bf16));
    bf16*  wb_c2   = (bf16*)carve((size_t)HID * HID * sizeof(bf16));
    bf16*  wb_comb = (bf16*)carve((size_t)HID * HID * sizeof(bf16));
    bf16*  wb_mot  = (bf16*)carve((size_t)NMOTIF * HID * sizeof(bf16));
    bf16*  wb_node = (bf16*)carve((size_t)NCLS * HID * sizeof(bf16));

    float* out       = (float*)d_out;
    float* out_edge  = out;                                       // [E, 64]
    float* out_motif = out + (size_t)E * 64;                      // [C, 512]
    float* out_node  = out + (size_t)E * 64 + (size_t)C * NMOTIF; // [N, 16]

    const int nTilesN = N >> 4;                                   // 12500 (N % 16 == 0)
    const int gemmGridN = nTilesN < GEMM_MAX_BLOCKS ? nTilesN : GEMM_MAX_BLOCKS;
    const int nTilesC = (C + 15) >> 4;                            // 16

    // 1) weight conversions (tiny)
    cvt_bf16_kernel<<<(HID * HID + 255) / 256, 256, 0, stream>>>(wb_lin, lin_w, HID * HID);
    cvt_bf16_kernel<<<(HID * HID + 255) / 256, 256, 0, stream>>>(wb_c1, conv1_w, HID * HID);
    cvt_bf16_kernel<<<(HID * HID + 255) / 256, 256, 0, stream>>>(wb_c2, conv2_w, HID * HID);
    build_wcomb_kernel<<<(HID * HID + 255) / 256, 256, 0, stream>>>(wb_comb, edge_w);
    cvt_bf16_kernel<<<(NMOTIF * HID + 255) / 256, 256, 0, stream>>>(wb_mot, motif_w, NMOTIF * HID);
    cvt_bf16_kernel<<<(NCLS * HID + 255) / 256, 256, 0, stream>>>(wb_node, node_w, NCLS * HID);

    // 2) project the 64-row relation table once (replaces the 600K-row edge_attr GEMM)
    rel_proj_kernel<<<NRELS, HID, 0, stream>>>(relp, rel_emb, lin_w, lin_b);

    // 3) x = node_emb[node_ids] @ lin_w.T + lin_b
    wmma_gemm_kernel<false, true, true, false><<<dim3(gemmGridN, 1), 256, 0, stream>>>(
        node_emb, nullptr, node_ids, wb_lin, lin_b, xbuf, N, HID);

    // 4) conv1: scatter messages then GEMM, x = relu((agg + x) @ W1.T + b1)
    zero_f32_kernel<<<4096, 256, 0, stream>>>(agg, (long)N * HID);
    edge_scatter_kernel<<<(E + 7) / 8, 256, 0, stream>>>(xbuf, relp, edge_index, rel_ids, agg, E);
    wmma_gemm_kernel<true, false, true, true><<<dim3(gemmGridN, 1), 256, 0, stream>>>(
        agg, xbuf, nullptr, wb_c1, conv1_b, xbuf, N, HID);

    // 5) conv2: x = (agg + x) @ W2.T + b2
    zero_f32_kernel<<<4096, 256, 0, stream>>>(agg, (long)N * HID);
    edge_scatter_kernel<<<(E + 7) / 8, 256, 0, stream>>>(xbuf, relp, edge_index, rel_ids, agg, E);
    wmma_gemm_kernel<true, false, true, false><<<dim3(gemmGridN, 1), 256, 0, stream>>>(
        agg, xbuf, nullptr, wb_c2, conv2_b, xbuf, N, HID);

    // 6) edge head: node-level projections P = x @ [W_src;W_dst].T (reuse agg buffer as P),
    //    then per-edge gather-add
    wmma_gemm_kernel<false, false, false, false><<<dim3(gemmGridN, 1), 256, 0, stream>>>(
        xbuf, nullptr, nullptr, wb_comb, nullptr, agg, N, HID);
    edge_head_kernel<<<(E + 7) / 8, 256, 0, stream>>>(agg, edge_index, edge_b, out_edge, E);

    // 7) motif head: gather centers, GEMM to 512 classes
    wmma_gemm_kernel<false, true, true, false>
        <<<dim3(nTilesC, (NMOTIF + 127) / 128), 256, 0, stream>>>(
        xbuf, nullptr, center_idx, wb_mot, motif_b, out_motif, C, NMOTIF);

    // 8) node classification head (Nout=16: only wave 0 computes; memory-bound anyway)
    wmma_gemm_kernel<false, false, true, false><<<dim3(gemmGridN, 1), 256, 0, stream>>>(
        xbuf, nullptr, nullptr, wb_node, node_b, out_node, N, NCLS);
}
